// HybridTransformerClassifier_65481071403822
// MI455X (gfx1250) — compile-verified, matrix operand order checked by reading, not run on hardware
//
#include <hip/hip_runtime.h>
#include <hip/hip_bf16.h>
#include <cstdint>

// ---------------------------------------------------------------------------
// Types for CDNA5 WMMA (wave32, 16x16x32 bf16 -> f32 accumulate)
// ---------------------------------------------------------------------------
typedef __bf16 bf16_t;
typedef __attribute__((ext_vector_type(16))) __bf16 v16bf;
typedef __attribute__((ext_vector_type(8)))  __bf16 v8bf;
typedef __attribute__((ext_vector_type(8)))  float  v8f;

#define EDIM 512
#define FDIM 2048
#define SEQ  2048
#define BATCH 8
#define LAYERS 6
#define NCLS 10
#define NQ 4
#define LN_EPS 1e-5f

// ---------------------------------------------------------------------------
// Tensor Data Mover support (gfx1250). Arity differs across toolchains:
// ROCm 7.2 clang-22 -> 5 args; therock/amdgpu-toolchain clang-23 -> 6 args
// (the latter is detectable because it ships amd_gfx1250_TDM.h).
// ---------------------------------------------------------------------------
#if defined(__has_builtin)
#if __has_builtin(__builtin_amdgcn_tensor_load_to_lds) && \
    __has_builtin(__builtin_amdgcn_s_wait_tensorcnt)
#define CDNA5_HAS_TDM 1
#endif
#endif
#ifndef CDNA5_HAS_TDM
#define CDNA5_HAS_TDM 0
#endif

#if __has_include(<hip/amd_detail/amd_gfx1250_TDM.h>)
#define CDNA5_TDM_ARITY6 1
#else
#define CDNA5_TDM_ARITY6 0
#endif

#if CDNA5_HAS_TDM
typedef __attribute__((ext_vector_type(4))) unsigned int v4u;
typedef __attribute__((ext_vector_type(8))) int          v8i;
typedef __attribute__((ext_vector_type(4))) int          v4i;

// Issue one TDM copy of a contiguous 16-row x kElems (bf16) tile into LDS.
// D# built per CDNA5 ISA 8.3/8.4: 2D tensor, data_size=2B, tile==tensor dims.
__device__ inline void tdm_load_a_tile(const bf16_t* gptr, uint32_t lds_byte_off,
                                       int kElems) {
    uint64_t ga = (uint64_t)(uintptr_t)gptr;
    v4u g0;
    g0[0] = 1u;                                        // count=1, user, no gather
    g0[1] = lds_byte_off;                              // lds_addr (bytes)
    g0[2] = (uint32_t)(ga & 0xFFFFFFFFu);              // global_addr[31:0]
    g0[3] = (uint32_t)((ga >> 32) & 0x01FFFFFFu)       // global_addr[56:32]
          | (2u << 30);                                // type = 2 ("image")
    uint32_t K = (uint32_t)kElems;
    v8i g1;
    g1[0] = (int)(1u << 16);                           // data_size=1 (2 bytes)
    g1[1] = (int)((K & 0xFFFFu) << 16);                // tensor_dim0[15:0] @63:48
    g1[2] = (int)(((K >> 16) & 0xFFFFu)                // tensor_dim0[31:16]
          | (16u << 16));                              // tensor_dim1 = 16 rows
    g1[3] = (int)((K & 0xFFFFu) << 16);                // tile_dim0 = K @127:112
    g1[4] = 16;                                        // tile_dim1 = 16 rows
    g1[5] = (int)K;                                    // tensor_dim0_stride[31:0]
    g1[6] = 0;                                         // stride hi / dim1_stride lo
    g1[7] = 0;
    v4i gz = {0, 0, 0, 0};
#if CDNA5_TDM_ARITY6
    v8i gz8 = {0, 0, 0, 0, 0, 0, 0, 0};
    __builtin_amdgcn_tensor_load_to_lds(g0, g1, gz, gz, gz8, 0);
#else
    __builtin_amdgcn_tensor_load_to_lds(g0, g1, gz, gz, 0);
#endif
}
#endif // CDNA5_HAS_TDM

// ---------------------------------------------------------------------------
// fp32 -> bf16 weight conversion (runs once per launch; weights then live in L2)
// ---------------------------------------------------------------------------
__global__ __launch_bounds__(256) void f2bf_kernel(const float* __restrict__ in,
                                                   bf16_t* __restrict__ out, int n) {
    int i = blockIdx.x * 256 + threadIdx.x;
    if (i < n) out[i] = (bf16_t)in[i];
}

// ---------------------------------------------------------------------------
// x = emb[tokens] + sinusoidal_pe   (fp32, M x E)
// ---------------------------------------------------------------------------
__global__ __launch_bounds__(256) void embed_kernel(const int* __restrict__ tokens,
                                                    const float* __restrict__ emb,
                                                    float* __restrict__ x) {
    int idx = blockIdx.x * 256 + threadIdx.x;          // < M*E = 2^23
    int e   = idx & (EDIM - 1);
    int row = idx >> 9;                                // token row in [0, B*S)
    int s   = row & (SEQ - 1);                         // position within sequence
    int tok = tokens[row];
    const float c = -9.210340371976184f / (float)EDIM; // -ln(10000)/E
    float i2   = (float)((e >> 1) << 1);
    float ang  = (float)s * __expf(i2 * c);
    float pe   = (e & 1) ? __cosf(ang) : __sinf(ang);
    x[idx] = emb[(size_t)tok * EDIM + e] + pe;
}

// ---------------------------------------------------------------------------
// attn_q = cos(x + theta) cast to bf16 (theta flattened over H*dk = E)
// ---------------------------------------------------------------------------
__global__ __launch_bounds__(256) void cosq_kernel(const float* __restrict__ x,
                                                   const float* __restrict__ theta,
                                                   bf16_t* __restrict__ out) {
    int idx = blockIdx.x * 256 + threadIdx.x;
    int e   = idx & (EDIM - 1);
    out[idx] = (bf16_t)__cosf(x[idx] + theta[e]);
}

// ---------------------------------------------------------------------------
// Fused GEMM + bias + residual + LayerNorm, templated on K.
//   A (M x K bf16): block's 16-row slab is TDM-staged into LDS once, then
//     fragments come from ds_load -- the global pipe is left to the streaming
//     W operand (N x K bf16, K contiguous, served out of the 192MB L2).
//   x (M x 512 fp32): residual in, LN output out, in place (block-private rows).
// Block: 256 threads = 8 wave32; each wave owns 4 adjacent 16x16 tiles so the
// block covers N=512 and the LayerNorm reduction is local.
// ---------------------------------------------------------------------------
template <int K>
__global__ __launch_bounds__(256) void gemm_ln_kernel(const bf16_t* __restrict__ A,
                                                      const bf16_t* __restrict__ W,
                                                      const float* __restrict__ bias,
                                                      const float* __restrict__ gamma,
                                                      const float* __restrict__ beta,
                                                      float* __restrict__ x) {
    __shared__ bf16_t ldsA[16 * K];
    __shared__ float  tile[16][EDIM + 4];

    const int m0    = blockIdx.x * 16;
    const int tid   = threadIdx.x;
    const int wave  = tid >> 5;          // 0..7
    const int lane  = tid & 31;
    const int half  = lane >> 4;         // ISA 16-bit fragment: K-half select
    const int lrow  = lane & 15;         // row (A) / column (B) within tile
    const int nBase = wave * 64;

    // ---- Stage the 16 x K A-slab (contiguous rows) into LDS --------------
#if CDNA5_HAS_TDM
    if (wave == 0) {
        tdm_load_a_tile(A + (size_t)m0 * K, (uint32_t)(uintptr_t)&ldsA[0], K);
        __builtin_amdgcn_s_wait_tensorcnt(0);
    }
#else
    {
        const v8bf* src = (const v8bf*)(A + (size_t)m0 * K);
        v8bf*       dst = (v8bf*)ldsA;
        for (int i = tid; i < 16 * K / 8; i += 256) dst[i] = src[i];
    }
#endif
    __syncthreads();

    const bf16_t* __restrict__ arow = ldsA + lrow * K;
    const bf16_t* __restrict__ wrow[4];
#pragma unroll
    for (int j = 0; j < 4; ++j)
        wrow[j] = W + (size_t)(nBase + 16 * j + lrow) * K;

    v8f acc[4];
#pragma unroll
    for (int j = 0; j < 4; ++j)
#pragma unroll
        for (int v = 0; v < 8; ++v) acc[j][v] = 0.0f;

    for (int k0 = 0; k0 < K; k0 += 32) {
        if (k0 + 32 < K)
            __builtin_prefetch(wrow[0] + k0 + 32, 0, 3);   // global_prefetch_b8

        // A fragment from LDS: lanes 0-15 hold row m, K {k0..+7, k0+16..+23};
        //                      lanes 16-31 hold K {k0+8..+15, k0+24..+31}.
        v8bf alo = *(const v8bf*)(arow + k0 + 8 * half);
        v8bf ahi = *(const v8bf*)(arow + k0 + 16 + 8 * half);
        v16bf afrag;
#pragma unroll
        for (int i = 0; i < 8; ++i) { afrag[i] = alo[i]; afrag[8 + i] = ahi[i]; }

        // Issue all W-fragment loads first so waits can be staged 3/2/1/0.
        v8bf blo[4], bhi[4];
#pragma unroll
        for (int j = 0; j < 4; ++j) {
            blo[j] = *(const v8bf*)(wrow[j] + k0 + 8 * half);
            bhi[j] = *(const v8bf*)(wrow[j] + k0 + 16 + 8 * half);
        }
#pragma unroll
        for (int j = 0; j < 4; ++j) {
            v16bf bfrag;
#pragma unroll
            for (int i = 0; i < 8; ++i) { bfrag[i] = blo[j][i]; bfrag[8 + i] = bhi[j][i]; }
            // D = A x B + C, f32 accumulate  (v_wmma_f32_16x16x32_bf16)
            acc[j] = __builtin_amdgcn_wmma_f32_16x16x32_bf16(
                false, afrag, false, bfrag, (short)0, acc[j], false, false);
        }
    }

    // Epilogue: C/D layout -> row = v + 8*(lane>=16), col = nBase+16j+(lane&15).
#pragma unroll
    for (int j = 0; j < 4; ++j) {
        int col = nBase + 16 * j + lrow;
        float bcol = bias[col];
#pragma unroll
        for (int v = 0; v < 8; ++v) {
            int row = v + 8 * half;
            tile[row][col] = acc[j][v] + bcol + x[(size_t)(m0 + row) * EDIM + col];
        }
    }
    __syncthreads();

    // LayerNorm: wave w normalizes rows 2w and 2w+1 via shuffle reductions.
#pragma unroll
    for (int rr = 0; rr < 2; ++rr) {
        int r = wave * 2 + rr;
        float s1 = 0.0f, s2 = 0.0f;
        for (int c = lane; c < EDIM; c += 32) {
            float v = tile[r][c];
            s1 += v; s2 += v * v;
        }
#pragma unroll
        for (int off = 16; off > 0; off >>= 1) {
            s1 += __shfl_xor(s1, off, 32);
            s2 += __shfl_xor(s2, off, 32);
        }
        float mu  = s1 * (1.0f / EDIM);
        float var = s2 * (1.0f / EDIM) - mu * mu;
        float inv = rsqrtf(var + LN_EPS);
        for (int c = lane; c < EDIM; c += 32) {
            x[(size_t)(m0 + r) * EDIM + c] =
                (tile[r][c] - mu) * inv * gamma[c] + beta[c];
        }
    }
}

// ---------------------------------------------------------------------------
// Quantum FFN front half: h = relu(lin1_W @ (cos(x[:, :4]) * cos(phi)) + b1)
// K = 4, so plain VALU; output bf16 to feed the lin2 WMMA GEMM.
// ---------------------------------------------------------------------------
__global__ __launch_bounds__(256) void ffn_h_kernel(const float* __restrict__ x,
                                                    const float* __restrict__ phi,
                                                    const float* __restrict__ W1,
                                                    const float* __restrict__ b1,
                                                    bf16_t* __restrict__ h) {
    int idx = blockIdx.x * 256 + threadIdx.x;   // < M*F = 2^25
    int f   = idx & (FDIM - 1);
    int m   = idx >> 11;
    const float* xr = x + (size_t)m * EDIM;
    float acc = b1[f];
#pragma unroll
    for (int q = 0; q < NQ; ++q)
        acc += __cosf(xr[q]) * __cosf(phi[q]) * W1[f * NQ + q];
    h[idx] = (bf16_t)fmaxf(acc, 0.0f);
}

// ---------------------------------------------------------------------------
// Mean pool over S, then (B,E) @ cls_W^T + cls_b. One block per batch element.
// ---------------------------------------------------------------------------
__global__ __launch_bounds__(256) void pool_cls_kernel(const float* __restrict__ x,
                                                       const float* __restrict__ clsW,
                                                       const float* __restrict__ clsb,
                                                       float* __restrict__ out) {
    __shared__ float pooled[EDIM];
    int b = blockIdx.x;
    int t = threadIdx.x;
    for (int e = t; e < EDIM; e += 256) {
        float s = 0.0f;
        const float* base = x + ((size_t)b * SEQ) * EDIM + e;
        for (int sIdx = 0; sIdx < SEQ; ++sIdx) s += base[(size_t)sIdx * EDIM];
        pooled[e] = s * (1.0f / SEQ);
    }
    __syncthreads();
    if (t < NCLS) {
        float acc = clsb[t];
        const float* wr = clsW + (size_t)t * EDIM;
        for (int e = 0; e < EDIM; ++e) acc += pooled[e] * wr[e];
        out[b * NCLS + t] = acc;
    }
}

// ---------------------------------------------------------------------------
// Host-side orchestration (graph-capture safe: only async launches on stream)
// ---------------------------------------------------------------------------
extern "C" void kernel_launch(void* const* d_in, const int* in_sizes, int n_in,
                              void* d_out, int out_size, void* d_ws, size_t ws_size,
                              hipStream_t stream) {
    (void)in_sizes; (void)n_in; (void)out_size; (void)ws_size;

    const int M = BATCH * SEQ;                  // 16384 token rows

    const int*   tokens    = (const int*)  d_in[0];
    const float* emb       = (const float*)d_in[1];
    const float* theta     = (const float*)d_in[2];   // (L, H, NQ) = (L, 512)
    const float* combine_W = (const float*)d_in[3];   // (L, E, E)
    const float* combine_b = (const float*)d_in[4];
    const float* ln1_g     = (const float*)d_in[5];
    const float* ln1_b     = (const float*)d_in[6];
    const float* phi       = (const float*)d_in[7];   // (L, NQ)
    const float* lin1_W    = (const float*)d_in[8];   // (L, F, NQ)
    const float* lin1_b    = (const float*)d_in[9];
    const float* lin2_W    = (const float*)d_in[10];  // (L, E, F)
    const float* lin2_b    = (const float*)d_in[11];
    const float* ln2_g     = (const float*)d_in[12];
    const float* ln2_b     = (const float*)d_in[13];
    const float* cls_W     = (const float*)d_in[14];
    const float* cls_b     = (const float*)d_in[15];

    // Workspace carve-up (256B aligned slices)
    auto align256 = [](size_t v) { return (v + 255) & ~(size_t)255; };
    char* ws = (char*)d_ws;
    size_t off = 0;
    float*  x    = (float*) (ws + off); off += align256((size_t)M * EDIM * 4);
    bf16_t* aq   = (bf16_t*)(ws + off); off += align256((size_t)M * EDIM * 2);
    bf16_t* hbuf = (bf16_t*)(ws + off); off += align256((size_t)M * FDIM * 2);
    bf16_t* Wc16 = (bf16_t*)(ws + off); off += align256((size_t)LAYERS * EDIM * EDIM * 2);
    bf16_t* W216 = (bf16_t*)(ws + off); off += align256((size_t)LAYERS * EDIM * FDIM * 2);

    // Weights -> bf16 (cheap; recomputed every call for determinism)
    {
        int n1 = LAYERS * EDIM * EDIM;
        f2bf_kernel<<<(n1 + 255) / 256, 256, 0, stream>>>(combine_W, Wc16, n1);
        int n2 = LAYERS * EDIM * FDIM;
        f2bf_kernel<<<(n2 + 255) / 256, 256, 0, stream>>>(lin2_W, W216, n2);
    }

    // Embedding + positional encoding
    embed_kernel<<<(M * EDIM) / 256, 256, 0, stream>>>(tokens, emb, x);

    for (int l = 0; l < LAYERS; ++l) {
        // Quantum attention: cos(x + theta) -> bf16
        cosq_kernel<<<(M * EDIM) / 256, 256, 0, stream>>>(x, theta + l * EDIM, aq);
        // combine GEMM (K = 512) + bias + residual + LN1, x updated in place
        gemm_ln_kernel<EDIM><<<M / 16, 256, 0, stream>>>(
            aq, Wc16 + (size_t)l * EDIM * EDIM, combine_b + l * EDIM,
            ln1_g + l * EDIM, ln1_b + l * EDIM, x);
        // Quantum FFN front half (K = 4, VALU) -> bf16 h
        ffn_h_kernel<<<((size_t)M * FDIM) / 256, 256, 0, stream>>>(
            x, phi + l * NQ, lin1_W + (size_t)l * FDIM * NQ, lin1_b + l * FDIM, hbuf);
        // lin2 GEMM (K = 2048) + bias + residual + LN2, x updated in place
        gemm_ln_kernel<FDIM><<<M / 16, 256, 0, stream>>>(
            hbuf, W216 + (size_t)l * EDIM * FDIM, lin2_b + l * EDIM,
            ln2_g + l * EDIM, ln2_b + l * EDIM, x);
    }

    // Mean pool + classifier
    pool_cls_kernel<<<BATCH, 256, 0, stream>>>(x, cls_W, cls_b, (float*)d_out);
}
